// HGTEvidenceModel_61194694034256
// MI455X (gfx1250) — compile-verified, matrix operand order checked by reading, not run on hardware
//
#include <hip/hip_runtime.h>
#include <stdint.h>
#include <stddef.h>

#define HID   256
#define HEADS 4
#define DHEAD 64
#define NTOT  66001
#define ETOT  476000

typedef __attribute__((ext_vector_type(16))) _Float16 v16h;
typedef __attribute__((ext_vector_type(8)))  float    v8f;

union FragAB { v16h v; uint32_t u[8]; };

__device__ __forceinline__ float gelu_tanh(float x) {
  float x3 = x * x * x;
  return 0.5f * x * (1.0f + tanhf(0.7978845608028654f * (x + 0.044715f * x3)));
}

__device__ __forceinline__ uint32_t pk2(float a, float b) {
  union { _Float16 h[2]; uint32_t u; } t;
  t.h[0] = (_Float16)a;
  t.h[1] = (_Float16)b;
  return t.u;
}

__device__ __forceinline__ unsigned fkey(float f) {
  unsigned u = __float_as_uint(f);
  return (u & 0x80000000u) ? ~u : (u | 0x80000000u);
}
__device__ __forceinline__ float fdecode(unsigned k) {
  return (k & 0x80000000u) ? __uint_as_float(k & 0x7fffffffu) : __uint_as_float(~k);
}

// ---------------------------------------------------------------------------
// Generic GEMM: Y[M,256] = act( f(X)[M,K] @ W[K,256] + bias ) (+ residual)
// TRANS: 0 = identity, 1 = gelu applied to X elements on load
// ACT:   0 = none, 1 = relu
// Block 256 threads = 8 waves; block tile 64x64; K chunked by 64.
// LDS tiles stored as packed f16 (uint32 pairs); all LDS traffic is b128.
// ---------------------------------------------------------------------------
template <int TRANS, int ACT>
__global__ void gemm256(const float* __restrict__ X, int M, int K,
                        const float* __restrict__ W, const float* __restrict__ Bv,
                        const float* __restrict__ R, float* __restrict__ Y) {
  // A: [row 0..63][k 0..63] f16 -> 64*32 uint32 (8 KB)
  // B: transposed [n 0..63][k 0..63] f16 -> 64*32 uint32 (8 KB)
  __shared__ __align__(16) uint32_t As[64 * 32];
  __shared__ __align__(16) uint32_t Bs[64 * 32];
  const int tid  = threadIdx.x;
  const int lane = tid & 31;
  const int wid  = tid >> 5;
  const int wrow = (wid >> 1) * 16;  // 0,16,32,48
  const int wcol = (wid & 1) * 32;   // 0,32
  const int m0 = blockIdx.x * 64;
  const int n0 = blockIdx.y * 64;

  v8f c0 = {}; v8f c1 = {};

  // A staging: thread owns row (tid>>2), 16 consecutive k at (tid&3)*16
  const int arow_ld = tid >> 2;
  const int ak_ld   = (tid & 3) * 16;
  // B staging: thread owns column (tid&63), 16 consecutive k at (tid>>6)*16
  const int bn_ld = tid & 63;
  const int bk_ld = (tid >> 6) * 16;

  for (int kc = 0; kc < K; kc += 64) {
    // --- stage A tile 64x64 (fp32 -> packed f16, optional gelu) ---
    {
      int gr = m0 + arow_ld;
      float tmp[16];
      if (gr < M) {
        const float* src = X + (size_t)gr * K + kc + ak_ld;
#pragma unroll
        for (int j = 0; j < 16; ++j) tmp[j] = src[j];
      } else {
#pragma unroll
        for (int j = 0; j < 16; ++j) tmp[j] = 0.0f;
      }
      if (TRANS == 1) {
#pragma unroll
        for (int j = 0; j < 16; ++j) tmp[j] = gelu_tanh(tmp[j]);
      }
      uint32_t p[8];
#pragma unroll
      for (int j = 0; j < 8; ++j) p[j] = pk2(tmp[2 * j], tmp[2 * j + 1]);
      uint32_t* dst = &As[arow_ld * 32 + (ak_ld >> 1)];
      *(uint4*)(dst)     = make_uint4(p[0], p[1], p[2], p[3]);
      *(uint4*)(dst + 4) = make_uint4(p[4], p[5], p[6], p[7]);
    }
    // --- stage B tile 64x64 transposed: Bs[n][k], coalesced global reads ---
    {
      float tmp[16];
      const float* src = W + (size_t)(kc + bk_ld) * HID + n0 + bn_ld;
#pragma unroll
      for (int j = 0; j < 16; ++j) tmp[j] = src[(size_t)j * HID];
      if (kc + 64 < K)
        __builtin_prefetch(W + (size_t)(kc + 64 + bk_ld) * HID + n0 + bn_ld, 0, 0);
      uint32_t p[8];
#pragma unroll
      for (int j = 0; j < 8; ++j) p[j] = pk2(tmp[2 * j], tmp[2 * j + 1]);
      uint32_t* dst = &Bs[bn_ld * 32 + (bk_ld >> 1)];
      *(uint4*)(dst)     = make_uint4(p[0], p[1], p[2], p[3]);
      *(uint4*)(dst + 4) = make_uint4(p[4], p[5], p[6], p[7]);
    }
    __syncthreads();

    // --- fragments per ISA 16-bit 16x32 layout; two k-subchunks of 32 ---
    const int kh4 = (lane >> 4) * 4;
    const uint32_t* ap  = &As[(wrow + (lane & 15)) * 32];
    const uint32_t* bp0 = &Bs[(wcol + (lane & 15)) * 32];
    const uint32_t* bp1 = &Bs[(wcol + 16 + (lane & 15)) * 32];
    FragAB fa0, fa1, fb00, fb01, fb10, fb11;
#pragma unroll
    for (int i = 0; i < 4; ++i) {
      fa0.u[i]      = ap[kh4 + i];
      fa0.u[4 + i]  = ap[8 + kh4 + i];
      fa1.u[i]      = ap[16 + kh4 + i];
      fa1.u[4 + i]  = ap[24 + kh4 + i];
      fb00.u[i]     = bp0[kh4 + i];
      fb00.u[4 + i] = bp0[8 + kh4 + i];
      fb01.u[i]     = bp0[16 + kh4 + i];
      fb01.u[4 + i] = bp0[24 + kh4 + i];
      fb10.u[i]     = bp1[kh4 + i];
      fb10.u[4 + i] = bp1[8 + kh4 + i];
      fb11.u[i]     = bp1[16 + kh4 + i];
      fb11.u[4 + i] = bp1[24 + kh4 + i];
    }
    c0 = __builtin_amdgcn_wmma_f32_16x16x32_f16(false, fa0.v, false, fb00.v, (short)0, c0, false, false);
    c1 = __builtin_amdgcn_wmma_f32_16x16x32_f16(false, fa0.v, false, fb10.v, (short)0, c1, false, false);
    c0 = __builtin_amdgcn_wmma_f32_16x16x32_f16(false, fa1.v, false, fb01.v, (short)0, c0, false, false);
    c1 = __builtin_amdgcn_wmma_f32_16x16x32_f16(false, fa1.v, false, fb11.v, (short)0, c1, false, false);
    __syncthreads();
  }

  // --- epilogue: D layout lane row = (lane>>4)*8 + r, col = lane&15 ---
  const int kh = lane >> 4;
#pragma unroll
  for (int r = 0; r < 8; ++r) {
    int row = m0 + wrow + kh * 8 + r;
    if (row >= M) continue;
    int col0 = n0 + wcol + (lane & 15);
    float v0 = c0[r] + Bv[col0];
    float v1 = c1[r] + Bv[col0 + 16];
    if (ACT == 1) { v0 = fmaxf(v0, 0.0f); v1 = fmaxf(v1, 0.0f); }
    if (R) {
      v0 += R[(size_t)row * HID + col0];
      v1 += R[(size_t)row * HID + col0 + 16];
    }
    Y[(size_t)row * HID + col0]      = v0;
    Y[(size_t)row * HID + col0 + 16] = v1;
  }
}

// ---------------------------------------------------------------------------
// Edge pass A: score[e,h] = (k[src] @ w_att[h]) . q[dst] * pri[h] / 8
// Wave per edge; lane = head(4) x f-slot(8 floats each). w_att staged in LDS.
// Also records global dst index and per-(dst,head) running max (uint key).
// ---------------------------------------------------------------------------
__global__ void edge_att(const float* __restrict__ karr, const float* __restrict__ qarr,
                         const float* __restrict__ watt, const float* __restrict__ pri,
                         const int* __restrict__ eidx, int E, int sb, int db,
                         float* __restrict__ score, int* __restrict__ gdst,
                         unsigned int* __restrict__ mxkey, int eoff) {
  __shared__ float ws[HEADS * DHEAD * DHEAD];  // 64 KB
  for (int i = threadIdx.x; i < HEADS * DHEAD * DHEAD; i += 256) ws[i] = watt[i];
  __syncthreads();

  int e = blockIdx.x * 8 + (threadIdx.x >> 5);
  int lane = threadIdx.x & 31;
  if (e >= E) return;

  int s = eidx[e] + sb;
  int d = eidx[E + e] + db;
  int h = lane >> 3;
  int fo = (lane & 7) * 8;

  float ks[8];
  const float* kp = karr + (size_t)s * HID + h * DHEAD + fo;
#pragma unroll
  for (int j = 0; j < 8; ++j) ks[j] = kp[j];

  float km[8] = {0.f, 0.f, 0.f, 0.f, 0.f, 0.f, 0.f, 0.f};
  const int gbase = lane & 24;  // first lane of this head's group
  for (int jd = 0; jd < 8; ++jd) {
#pragma unroll
    for (int dd = 0; dd < 8; ++dd) {
      float kv = __shfl(ks[dd], gbase + jd, 32);
      const float* wrow = ws + h * DHEAD * DHEAD + (jd * 8 + dd) * DHEAD + fo;
#pragma unroll
      for (int j = 0; j < 8; ++j) km[j] = fmaf(kv, wrow[j], km[j]);
    }
  }
  const float* qp = qarr + (size_t)d * HID + h * DHEAD + fo;
  float sc = 0.f;
#pragma unroll
  for (int j = 0; j < 8; ++j) sc = fmaf(km[j], qp[j], sc);
  sc += __shfl_xor(sc, 1, 32);
  sc += __shfl_xor(sc, 2, 32);
  sc += __shfl_xor(sc, 4, 32);
  sc *= pri[h] * 0.125f;
  if ((lane & 7) == 0) {
    score[(size_t)(eoff + e) * HEADS + h] = sc;
    atomicMax(&mxkey[d * HEADS + h], fkey(sc));
  }
  if (lane == 0) gdst[eoff + e] = d;
}

// ---------------------------------------------------------------------------
// Edge pass B: ex = exp(score - max[dst,h]); den[dst,h] += ex; score <- ex
// ---------------------------------------------------------------------------
__global__ void softmax_norm(const int* __restrict__ gdst, const unsigned int* __restrict__ mxkey,
                             float* __restrict__ score, float* __restrict__ den, int total) {
  int t = blockIdx.x * 256 + threadIdx.x;
  if (t >= total) return;
  int e = t >> 2, h = t & 3;
  int d = gdst[e];
  float mx = fdecode(mxkey[d * HEADS + h]);
  float ex = __expf(score[t] - mx);
  score[t] = ex;
  atomicAdd(&den[d * HEADS + h], ex);
}

// ---------------------------------------------------------------------------
// Edge pass C: agg[dst] += attn * (v[src] @ w_msg[h]); attn = ex/max(den,1e-9)
// ---------------------------------------------------------------------------
__global__ void edge_msg(const float* __restrict__ varr, const float* __restrict__ wmsg,
                         const int* __restrict__ eidx, int E, int sb, int db,
                         const float* __restrict__ score, const float* __restrict__ den,
                         float* __restrict__ agg, int eoff) {
  __shared__ float ws[HEADS * DHEAD * DHEAD];  // 64 KB
  for (int i = threadIdx.x; i < HEADS * DHEAD * DHEAD; i += 256) ws[i] = wmsg[i];
  __syncthreads();

  int e = blockIdx.x * 8 + (threadIdx.x >> 5);
  int lane = threadIdx.x & 31;
  if (e >= E) return;

  int s = eidx[e] + sb;
  int d = eidx[E + e] + db;
  int h = lane >> 3;
  int fo = (lane & 7) * 8;

  float vs[8];
  const float* vp = varr + (size_t)s * HID + h * DHEAD + fo;
#pragma unroll
  for (int j = 0; j < 8; ++j) vs[j] = vp[j];

  float mm[8] = {0.f, 0.f, 0.f, 0.f, 0.f, 0.f, 0.f, 0.f};
  const int gbase = lane & 24;
  for (int jd = 0; jd < 8; ++jd) {
#pragma unroll
    for (int dd = 0; dd < 8; ++dd) {
      float vv = __shfl(vs[dd], gbase + jd, 32);
      const float* wrow = ws + h * DHEAD * DHEAD + (jd * 8 + dd) * DHEAD + fo;
#pragma unroll
      for (int j = 0; j < 8; ++j) mm[j] = fmaf(vv, wrow[j], mm[j]);
    }
  }
  float ex = score[(size_t)(eoff + e) * HEADS + h];
  float dv = den[d * HEADS + h];
  float attn = ex / fmaxf(dv, 1e-9f);
  float* ap = agg + (size_t)d * HID + h * DHEAD + fo;
#pragma unroll
  for (int j = 0; j < 8; ++j) atomicAdd(&ap[j], mm[j] * attn);
}

// ---------------------------------------------------------------------------
// Final heads: wq = qh @ W (1x256), then out[row] = wq . h[row]
// ---------------------------------------------------------------------------
__global__ void vecmat256(const float* __restrict__ qh, const float* __restrict__ W,
                          float* __restrict__ wq) {
  int c = threadIdx.x;
  float s = 0.f;
  for (int k = 0; k < HID; ++k) s = fmaf(qh[k], W[k * HID + c], s);
  wq[c] = s;
}

__global__ void rowscore(const float* __restrict__ harr, int base, int n,
                         const float* __restrict__ wq, float* __restrict__ out) {
  int row = blockIdx.x * 8 + (threadIdx.x >> 5);
  int lane = threadIdx.x & 31;
  if (row >= n) return;
  const float* hp = harr + (size_t)(base + row) * HID;
  float s = 0.f;
#pragma unroll
  for (int j = 0; j < 8; ++j) s = fmaf(hp[lane + j * 32], wq[lane + j * 32], s);
  s += __shfl_xor(s, 1, 32);
  s += __shfl_xor(s, 2, 32);
  s += __shfl_xor(s, 4, 32);
  s += __shfl_xor(s, 8, 32);
  s += __shfl_xor(s, 16, 32);
  if (lane == 0) out[row] = s;
}

// ---------------------------------------------------------------------------
// Host-side orchestration
// ---------------------------------------------------------------------------
extern "C" void kernel_launch(void* const* d_in, const int* in_sizes, int n_in,
                              void* d_out, int out_size, void* d_ws, size_t ws_size,
                              hipStream_t stream) {
  (void)in_sizes; (void)n_in; (void)out_size; (void)ws_size;

  // Node types: query, textblock, table, cell, image, caption
  static const int NB[6] = {0, 1, 16001, 18001, 50001, 58001};
  static const int NN[6] = {1, 16000, 2000, 32000, 8000, 8000};
  // 18 relations: src type, dst type, count, cumulative offset
  static const int ES[18] = {0,0,0,0, 1,2,3,4, 1,2,1,4, 2,3,3,3, 4,5};
  static const int ED[18] = {1,2,3,4, 0,0,0,0, 2,1,4,1, 3,2,3,3, 5,4};
  static const int EN[18] = {16000,2000,32000,8000, 16000,2000,32000,8000,
                             4000,4000,8000,8000, 32000,32000,128000,128000, 8000,8000};
  static const int EO[18] = {0,16000,18000,50000, 58000,74000,76000,108000,
                             116000,120000,124000,132000, 140000,172000,204000,332000,
                             460000,468000};

  // Input index map (setup_inputs insertion order):
  //   x_dict[6] -> 0..5 ; edges[18] -> 6..23 ; input_proj w,b x6 -> 24..35
  //   layer l base L=36+l*102: k(12) q(12) v(12) a(12) w_att(18) w_msg(18) pri(18)
  //   heads[5] -> 240..244
  const int IP = 24;

  // Workspace carve
  size_t off = 0;
  auto carve = [&](size_t bytes) -> char* {
    char* p = (char*)d_ws + off;
    off = (off + bytes + 255) & ~(size_t)255;
    return p;
  };
  const size_t szH = (size_t)NTOT * HID * sizeof(float);
  float*    hA    = (float*)carve(szH);
  float*    hB    = (float*)carve(szH);
  float*    Karr  = (float*)carve(szH);
  float*    Qarr  = (float*)carve(szH);
  float*    Varr  = (float*)carve(szH);
  float*    Agg   = (float*)carve(szH);
  float*    score = (float*)carve((size_t)ETOT * HEADS * sizeof(float));
  int*      gdst  = (int*)carve((size_t)ETOT * sizeof(int));
  unsigned* mxkey = (unsigned*)carve((size_t)NTOT * HEADS * sizeof(unsigned));
  float*    den   = (float*)carve((size_t)NTOT * HEADS * sizeof(float));
  float*    wq    = (float*)carve(5 * HID * sizeof(float));

  // 1) Input projection + ReLU: h = relu(X @ Wip + b)
  for (int nt = 0; nt < 6; ++nt) {
    dim3 grid((NN[nt] + 63) / 64, 4);
    gemm256<0, 1><<<grid, 256, 0, stream>>>(
        (const float*)d_in[nt], NN[nt], 4096,
        (const float*)d_in[IP + 2 * nt], (const float*)d_in[IP + 2 * nt + 1],
        nullptr, hA + (size_t)NB[nt] * HID);
  }

  // 2) HGT layers
  float* cur = hA;
  float* nxt = hB;
  for (int l = 0; l < 2; ++l) {
    const int L = 36 + l * 102;
    hipMemsetAsync(mxkey, 0, (size_t)NTOT * HEADS * sizeof(unsigned), stream);
    hipMemsetAsync(den,   0, (size_t)NTOT * HEADS * sizeof(float), stream);
    hipMemsetAsync(Agg,   0, szH, stream);

    // K/Q/V projections
    for (int nt = 0; nt < 6; ++nt) {
      dim3 grid((NN[nt] + 63) / 64, 4);
      const float* x = cur + (size_t)NB[nt] * HID;
      gemm256<0, 0><<<grid, 256, 0, stream>>>(
          x, NN[nt], HID, (const float*)d_in[L + 2 * nt],
          (const float*)d_in[L + 2 * nt + 1], nullptr, Karr + (size_t)NB[nt] * HID);
      gemm256<0, 0><<<grid, 256, 0, stream>>>(
          x, NN[nt], HID, (const float*)d_in[L + 12 + 2 * nt],
          (const float*)d_in[L + 12 + 2 * nt + 1], nullptr, Qarr + (size_t)NB[nt] * HID);
      gemm256<0, 0><<<grid, 256, 0, stream>>>(
          x, NN[nt], HID, (const float*)d_in[L + 24 + 2 * nt],
          (const float*)d_in[L + 24 + 2 * nt + 1], nullptr, Varr + (size_t)NB[nt] * HID);
    }

    // Pass A: scores + segment max
    for (int r = 0; r < 18; ++r) {
      edge_att<<<(EN[r] + 7) / 8, 256, 0, stream>>>(
          Karr, Qarr, (const float*)d_in[L + 48 + r], (const float*)d_in[L + 84 + r],
          (const int*)d_in[6 + r], EN[r], NB[ES[r]], NB[ED[r]],
          score, gdst, mxkey, EO[r]);
    }
    // Pass B: exp + segment sum
    softmax_norm<<<(ETOT * HEADS + 255) / 256, 256, 0, stream>>>(
        gdst, mxkey, score, den, ETOT * HEADS);
    // Pass C: messages + scatter-add
    for (int r = 0; r < 18; ++r) {
      edge_msg<<<(EN[r] + 7) / 8, 256, 0, stream>>>(
          Varr, (const float*)d_in[L + 66 + r],
          (const int*)d_in[6 + r], EN[r], NB[ES[r]], NB[ED[r]],
          score, den, Agg, EO[r]);
    }
    // Output projection: next = cur + A(gelu(agg))
    for (int nt = 0; nt < 6; ++nt) {
      dim3 grid((NN[nt] + 63) / 64, 4);
      gemm256<1, 0><<<grid, 256, 0, stream>>>(
          Agg + (size_t)NB[nt] * HID, NN[nt], HID,
          (const float*)d_in[L + 36 + 2 * nt], (const float*)d_in[L + 36 + 2 * nt + 1],
          cur + (size_t)NB[nt] * HID, nxt + (size_t)NB[nt] * HID);
    }
    float* t = cur; cur = nxt; nxt = t;
  }

  // 3) Bilinear heads. SCORED order: textblock, cell, image, caption, table
  static const int SIDX[5] = {1, 3, 4, 5, 2};
  static const int OOFF[5] = {0, 16000, 48000, 56000, 64000};
  for (int j = 0; j < 5; ++j) {
    vecmat256<<<1, 256, 0, stream>>>(cur /* query is global row 0 */,
                                     (const float*)d_in[240 + j], wq + j * HID);
    rowscore<<<(NN[SIDX[j]] + 7) / 8, 256, 0, stream>>>(
        cur, NB[SIDX[j]], NN[SIDX[j]], wq + j * HID, (float*)d_out + OOFF[j]);
  }
}
